// StockMixerWithOutIndicatorMixing_38568806318352
// MI455X (gfx1250) — compile-verified, matrix operand order checked by reading, not run on hardware
//
#include <hip/hip_runtime.h>
#include <hip/hip_bf16.h>
#include <stdint.h>

typedef __attribute__((ext_vector_type(16))) __bf16 v16bf;
typedef __attribute__((ext_vector_type(8)))  float  v8f;
typedef __attribute__((ext_vector_type(4)))  int    v4i;
typedef __bf16 bf16;

#define NSTK   4096
#define MIX    224
#define MARKET 64
#define HTOT   (NSTK * MIX)   // 917504
#define RED_BLOCKS 512
#define XELEMS (NSTK * 128 * 128)   // 67108864

#if defined(__HIP_DEVICE_COMPILE__) && __has_builtin(__builtin_amdgcn_global_load_async_to_lds_b128)
#define HAVE_ASYNC_LDS 1
#else
#define HAVE_ASYNC_LDS 0
#endif

__device__ __forceinline__ bf16 f2bf(float f) {
    union { float f; uint32_t u; } a; a.f = f;
    uint32_t r = a.u + 0x7FFFu + ((a.u >> 16) & 1u);   // round-to-nearest-even
    uint16_t h = (uint16_t)(r >> 16);
    union { uint16_t s; bf16 b; } o; o.s = h;
    return o.b;
}

__device__ __forceinline__ float hswish(float v) {
    return v * fminf(fmaxf(v + 3.0f, 0.0f), 6.0f) * (1.0f / 6.0f);
}

// Load one 16x32 bf16 WMMA operand fragment.
// Layout (ISA 7.12.2): lanes 0-15 hold rows/cols 0-15 with K = {0..7,16..23},
// lanes 16-31 hold K = {8..15,24..31}: two K-contiguous 8-element (16B) runs.
__device__ __forceinline__ v16bf ld_frag(const bf16* __restrict__ base,
                                         int row, int stride, int kb) {
    const bf16* p = base + row * stride + kb;
    union { v16bf v; uint4 q[2]; } u;
    u.q[0] = *(const uint4*)(p);
    u.q[1] = *(const uint4*)(p + 16);
    return u.v;
}

// ---------------- prep kernels ----------------

// x (fp32) -> bf16, whole tensor (makes working set L2-resident: 134MB < 192MB L2)
__global__ void prep_xbf16(const float* __restrict__ x, bf16* __restrict__ xo) {
    for (int i4 = blockIdx.x * blockDim.x + threadIdx.x; i4 < XELEMS / 4;
         i4 += gridDim.x * blockDim.x) {
        float4 v = ((const float4*)x)[i4];
        union { bf16 b[4]; uint2 u; } pk;
        pk.b[0] = f2bf(v.x); pk.b[1] = f2bf(v.y);
        pk.b[2] = f2bf(v.z); pk.b[3] = f2bf(v.w);
        *(uint2*)(xo + 4 * (size_t)i4) = pk.u;
    }
}

// wt[o][kk*128 + i] = cw[o,i,kk]   (bf16, K-major "B-transposed" layout)
__global__ void prep_convw(const float* __restrict__ cw, bf16* __restrict__ wt, int k) {
    int n = 128 * 128 * k;
    for (int d = blockIdx.x * blockDim.x + threadIdx.x; d < n; d += gridDim.x * blockDim.x) {
        int o  = d / (128 * k);
        int rr = d % (128 * k);
        int kk = rr >> 7;
        int i  = rr & 127;
        wt[d] = f2bf(cw[(o * 128 + i) * k + kk]);
    }
}

// out = tril(L) in bf16 (row-major tk x tk)
__global__ void prep_tril(const float* __restrict__ L, bf16* __restrict__ out, int tk) {
    int n = tk * tk;
    for (int d = blockIdx.x * blockDim.x + threadIdx.x; d < n; d += gridDim.x * blockDim.x) {
        int t = d / tk, s = d % tk;
        out[d] = f2bf(s <= t ? L[d] : 0.0f);
    }
}

// ---------------- fused per-scale kernel ----------------
// One block per sample. conv GEMM -> LN(tk,F) -> u=hsw(TriL1@h) -> v=TriL2@u ->
// xt+=v -> fc head into h_cat[:, col_off:col_off+tk].
template<int KF>
__global__ void __launch_bounds__(256)
fused_scale_kernel(const float* __restrict__ x,     // fp32 fallback source
                   const bf16*  __restrict__ xg16,  // bf16 source (may be null)
                   const bf16*  __restrict__ wt,    // (128, 128*KF) K-major
                   const float* __restrict__ cb,    // (128)
                   const float* __restrict__ lg,    // (tk,128)
                   const float* __restrict__ lb,    // (tk,128)
                   const bf16*  __restrict__ tl1,   // (tk,tk) bf16 tril
                   const bf16*  __restrict__ tl2,   // (tk,tk)
                   const float* __restrict__ fcw,   // (128)
                   const float* __restrict__ fcb,   // (1)
                   float* __restrict__ hc,          // (4096,224)
                   int col_off) {
    constexpr int TK = 128 / KF;
    constexpr int KD = 128 * KF;
    const int n    = blockIdx.x;
    const int tid  = threadIdx.x;
    const int lane = tid & 31;
    const int wv   = tid >> 5;
    const int rl   = lane & 15;
    const int sel  = (lane & 16) ? 8 : 0;   // K sub-select for A/B frags
    const int rsel = (lane & 16) ? 8 : 0;   // row sub-select for C/D

    extern __shared__ __align__(16) char smem[];
    bf16*  xbf = (bf16*)smem;                               // 32768 B (x[n] bf16; later aliased as uT)
    float* xt  = (float*)(smem + 32768);                    // TK*128*4
    bf16*  hT  = (bf16*)(smem + 32768 + TK * 128 * 4);      // 128*TK*2 (h transposed, f-major)
    float* scr = (float*)(smem + 32768 + TK * 128 * 4 + TK * 128 * 2); // 512 floats

    // ---- stage x[n] (128x128) as bf16 in LDS ----
    if (xg16) {
        const bf16* src = xg16 + (size_t)n * 16384;
#if HAVE_ASYNC_LDS
        // CDNA5 async copy: global -> LDS without VGPR round trip (ASYNCcnt)
        for (int i = tid; i < 2048; i += 256) {
            __builtin_amdgcn_global_load_async_to_lds_b128(
                (__attribute__((address_space(1))) v4i*)(src + i * 8),
                (__attribute__((address_space(3))) v4i*)(xbf + i * 8),
                0, 0);
        }
#if __has_builtin(__builtin_amdgcn_s_wait_asynccnt)
        __builtin_amdgcn_s_wait_asynccnt(0);
#else
        asm volatile("s_wait_asynccnt 0x0" ::: "memory");
#endif
#else
        for (int i = tid; i < 2048; i += 256)
            *(uint4*)(xbf + i * 8) = *(const uint4*)(src + i * 8);
#endif
    } else {
        const float* xg = x + (size_t)n * 16384;
        for (int i4 = tid; i4 < 4096; i4 += 256) {
            float4 v = ((const float4*)xg)[i4];
            union { bf16 b[4]; uint2 u; } pk;
            pk.b[0] = f2bf(v.x); pk.b[1] = f2bf(v.y);
            pk.b[2] = f2bf(v.z); pk.b[3] = f2bf(v.w);
            *(uint2*)(xbf + 4 * i4) = pk.u;
        }
    }
    __syncthreads();

    // ---- GEMM1: xt(tk x 128) = A(tk x KD = x view) @ wt^T, + bias ----
    for (int tile = wv; tile < (TK / 16) * 8; tile += 8) {
        int mt = tile >> 3, nt = tile & 7;
        v8f acc = {};
        #pragma unroll
        for (int ks = 0; ks < KD / 32; ++ks) {
            int kb = ks * 32 + sel;
            v16bf a = ld_frag(xbf, mt * 16 + rl, KD, kb);
            v16bf b = ld_frag(wt,  nt * 16 + rl, KD, kb);
            acc = __builtin_amdgcn_wmma_f32_16x16x32_bf16(false, a, false, b,
                                                          (short)0, acc, false, false);
        }
        int col = nt * 16 + rl;
        float bias = cb[col];
        int m0 = mt * 16 + rsel;
        #pragma unroll
        for (int j = 0; j < 8; ++j) xt[(m0 + j) * 128 + col] = acc[j] + bias;
    }
    __syncthreads();

    // ---- LayerNorm over the whole (TK,128) slab ----
    float s = 0.f, s2 = 0.f;
    for (int i = tid; i < TK * 128; i += 256) { float v = xt[i]; s += v; s2 += v * v; }
    scr[tid] = s; scr[256 + tid] = s2;
    __syncthreads();
    for (int off = 128; off > 0; off >>= 1) {
        if (tid < off) { scr[tid] += scr[tid + off]; scr[256 + tid] += scr[256 + tid + off]; }
        __syncthreads();
    }
    const float inv = 1.0f / (float)(TK * 128);
    float mu  = scr[0] * inv;
    float var = scr[256] * inv - mu * mu;
    float rs  = rsqrtf(var + 1e-5f);
    __syncthreads();

    // ---- h = LN(xt)*lg + lb, stored transposed bf16 (f-major) ----
    for (int i = tid; i < TK * 128; i += 256) {
        int t = i >> 7, f = i & 127;
        float v = (xt[i] - mu) * rs * lg[i] + lb[i];
        hT[f * TK + t] = f2bf(v);
    }
    __syncthreads();

    bf16* uT = xbf;   // reuse x buffer (x no longer needed)

    // ---- GEMM2: u = hswish(TriL1 @ h), triangular K-skip ----
    for (int tile = wv; tile < (TK / 16) * 8; tile += 8) {
        int mt = tile >> 3, nt = tile & 7;
        v8f acc = {};
        int ksteps = (mt * 16 + 15) / 32 + 1;   // only K tiles touching the lower triangle
        for (int ks = 0; ks < ksteps; ++ks) {
            int kb = ks * 32 + sel;
            v16bf a = ld_frag(tl1, mt * 16 + rl, TK, kb);
            v16bf b = ld_frag(hT,  nt * 16 + rl, TK, kb);
            acc = __builtin_amdgcn_wmma_f32_16x16x32_bf16(false, a, false, b,
                                                          (short)0, acc, false, false);
        }
        int f  = nt * 16 + rl;
        int m0 = mt * 16 + rsel;
        #pragma unroll
        for (int j = 0; j < 8; ++j) uT[f * TK + (m0 + j)] = f2bf(hswish(acc[j]));
    }
    __syncthreads();

    // ---- GEMM3: v = TriL2 @ u ; xt += v ----
    for (int tile = wv; tile < (TK / 16) * 8; tile += 8) {
        int mt = tile >> 3, nt = tile & 7;
        v8f acc = {};
        int ksteps = (mt * 16 + 15) / 32 + 1;
        for (int ks = 0; ks < ksteps; ++ks) {
            int kb = ks * 32 + sel;
            v16bf a = ld_frag(tl2, mt * 16 + rl, TK, kb);
            v16bf b = ld_frag(uT,  nt * 16 + rl, TK, kb);
            acc = __builtin_amdgcn_wmma_f32_16x16x32_bf16(false, a, false, b,
                                                          (short)0, acc, false, false);
        }
        int col = nt * 16 + rl;
        int m0  = mt * 16 + rsel;
        #pragma unroll
        for (int j = 0; j < 8; ++j) xt[(m0 + j) * 128 + col] += acc[j];
    }
    __syncthreads();

    // ---- fc head: part[n,t] = xt[t,:] . fcw + fcb ----
    if (tid < 128) scr[tid] = fcw[tid];
    __syncthreads();
    if (tid < TK) {
        float a = fcb[0];
        #pragma unroll 4
        for (int f = 0; f < 128; ++f) a += xt[tid * 128 + f] * scr[f];
        hc[(size_t)n * MIX + col_off + tid] = a;
    }
}

// ---------------- stock-mixing stage (scalar; ~0.3% of FLOPs) ----------------

__global__ void reduce1(const float* __restrict__ hc, float* __restrict__ part) {
    __shared__ float r1[256], r2[256];
    int tid = threadIdx.x;
    float s = 0.f, s2 = 0.f;
    for (int i = blockIdx.x * 256 + tid; i < HTOT; i += RED_BLOCKS * 256) {
        float v = hc[i]; s += v; s2 += v * v;
    }
    r1[tid] = s; r2[tid] = s2;
    __syncthreads();
    for (int off = 128; off > 0; off >>= 1) {
        if (tid < off) { r1[tid] += r1[tid + off]; r2[tid] += r2[tid + off]; }
        __syncthreads();
    }
    if (tid == 0) { part[2 * blockIdx.x] = r1[0]; part[2 * blockIdx.x + 1] = r2[0]; }
}

__global__ void reduce2(const float* __restrict__ part, float* __restrict__ stats) {
    __shared__ float r1[256], r2[256];
    int tid = threadIdx.x;
    float s = 0.f, s2 = 0.f;
    for (int i = tid; i < RED_BLOCKS; i += 256) { s += part[2 * i]; s2 += part[2 * i + 1]; }
    r1[tid] = s; r2[tid] = s2;
    __syncthreads();
    for (int off = 128; off > 0; off >>= 1) {
        if (tid < off) { r1[tid] += r1[tid + off]; r2[tid] += r2[tid + off]; }
        __syncthreads();
    }
    if (tid == 0) {
        float mu  = r1[0] / (float)HTOT;
        float var = r2[0] / (float)HTOT - mu * mu;
        stats[0] = mu;
        stats[1] = rsqrtf(var + 1e-5f);
    }
}

// g[m][j] = hswish( sum_s M1[m,s] * hn[s,j] ), one block per column j
__global__ void mix1_kernel(const float* __restrict__ hc, const float* __restrict__ stats,
                            const float* __restrict__ slg, const float* __restrict__ slb,
                            const float* __restrict__ M1, float* __restrict__ g) {
    __shared__ float col[NSTK];
    __shared__ float part[256];
    int j = blockIdx.x, tid = threadIdx.x;
    float mu = stats[0], rs = stats[1];
    for (int s = tid; s < NSTK; s += 256) {
        size_t idx = (size_t)s * MIX + j;
        col[s] = (hc[idx] - mu) * rs * slg[idx] + slb[idx];
    }
    __syncthreads();
    int m = tid & 63, c = tid >> 6;
    float p = 0.f;
    for (int s = c * (NSTK / 4); s < (c + 1) * (NSTK / 4); ++s)
        p += M1[(size_t)m * NSTK + s] * col[s];
    part[tid] = p;
    __syncthreads();
    if (tid < 64) {
        float v = part[tid] + part[tid + 64] + part[tid + 128] + part[tid + 192];
        g[tid * MIX + j] = hswish(v);
    }
}

// out[n] = sum_j h[n,j]*(w[j]+w[MIX+j]) + z[n,j]*w[MIX+j]  + b,  z = M2 @ g
__global__ void mix2_kernel(const float* __restrict__ hc, const float* __restrict__ g,
                            const float* __restrict__ M2, const float* __restrict__ fw,
                            const float* __restrict__ fb, float* __restrict__ out) {
    __shared__ float gs[MARKET * MIX];   // 57344 B: whole g matrix, coalesced staging
    __shared__ float m2row[MARKET];
    __shared__ float red[256];
    int n = blockIdx.x, tid = threadIdx.x;
    if (tid < MARKET) m2row[tid] = M2[(size_t)n * MARKET + tid];
    for (int i = tid; i < MARKET * MIX; i += 256) gs[i] = g[i];
    __syncthreads();
    float contrib = 0.f;
    if (tid < MIX) {
        float z = 0.f;
        #pragma unroll 8
        for (int m = 0; m < MARKET; ++m) z += m2row[m] * gs[m * MIX + tid];
        float hv = hc[(size_t)n * MIX + tid];
        float w1 = fw[tid], w2 = fw[MIX + tid];
        contrib = hv * (w1 + w2) + z * w2;
    }
    red[tid] = contrib;
    __syncthreads();
    for (int off = 128; off > 0; off >>= 1) {
        if (tid < off) red[tid] += red[tid + off];
        __syncthreads();
    }
    if (tid == 0) out[n] = red[0] + fb[0];
}

// ---------------- launch ----------------

extern "C" void kernel_launch(void* const* d_in, const int* in_sizes, int n_in,
                              void* d_out, int out_size, void* d_ws, size_t ws_size,
                              hipStream_t stream) {
    const float* x    = (const float*)d_in[0];
    const float* cw[3]  = { (const float*)d_in[1],  (const float*)d_in[7],  (const float*)d_in[13] };
    const float* cb[3]  = { (const float*)d_in[2],  (const float*)d_in[8],  (const float*)d_in[14] };
    const float* lg[3]  = { (const float*)d_in[3],  (const float*)d_in[9],  (const float*)d_in[15] };
    const float* lb[3]  = { (const float*)d_in[4],  (const float*)d_in[10], (const float*)d_in[16] };
    const float* L1[3]  = { (const float*)d_in[5],  (const float*)d_in[11], (const float*)d_in[17] };
    const float* L2[3]  = { (const float*)d_in[6],  (const float*)d_in[12], (const float*)d_in[18] };
    const float* fcw    = (const float*)d_in[19];
    const float* fcb    = (const float*)d_in[20];
    const float* sm_lg  = (const float*)d_in[21];
    const float* sm_lb  = (const float*)d_in[22];
    const float* M1     = (const float*)d_in[23];
    const float* M2     = (const float*)d_in[24];
    const float* sm_fcw = (const float*)d_in[25];
    const float* sm_fcb = (const float*)d_in[26];
    float* out = (float*)d_out;

    // workspace layout (bytes, 16B-aligned)
    char* ws = (char*)d_ws;
    float* hc       = (float*)(ws + 0);                       // 4096*224*4 = 3,670,016
    size_t off = 3670016;
    bf16* wt[3];
    const int kf[3] = {1, 2, 4};
    for (int si = 0; si < 3; ++si) { wt[si] = (bf16*)(ws + off); off += (size_t)128 * 128 * kf[si] * 2; }
    bf16* tl1[3]; bf16* tl2[3];
    for (int si = 0; si < 3; ++si) {
        int tk = 128 / kf[si];
        tl1[si] = (bf16*)(ws + off); off += (size_t)tk * tk * 2;
        tl2[si] = (bf16*)(ws + off); off += (size_t)tk * tk * 2;
    }
    off = (off + 255) & ~(size_t)255;
    float* partials = (float*)(ws + off); off += RED_BLOCKS * 2 * 4;
    off = (off + 255) & ~(size_t)255;
    float* stats    = (float*)(ws + off); off += 256;
    float* gbuf     = (float*)(ws + off); off += (size_t)MARKET * MIX * 4;
    off = (off + 255) & ~(size_t)255;

    // Optional bf16 copy of x (134 MB): fits MI455X's 192MB L2, halves HBM traffic
    // for the three per-scale kernels. Falls back if workspace is too small.
    bf16* xg16 = nullptr;
    if (ws_size >= off + (size_t)XELEMS * 2) {
        xg16 = (bf16*)(ws + off);
        prep_xbf16<<<32768, 256, 0, stream>>>(x, xg16);
    }

    // ---- prep: bf16 K-major conv weights + bf16 tril factors ----
    for (int si = 0; si < 3; ++si) {
        int nelem = 128 * 128 * kf[si];
        prep_convw<<<(nelem + 255) / 256, 256, 0, stream>>>(cw[si], wt[si], kf[si]);
        int tk = 128 / kf[si];
        prep_tril<<<(tk * tk + 255) / 256, 256, 0, stream>>>(L1[si], tl1[si], tk);
        prep_tril<<<(tk * tk + 255) / 256, 256, 0, stream>>>(L2[si], tl2[si], tk);
    }

    // ---- fused per-scale WMMA kernels ----
    {
        constexpr int TK = 128;
        size_t sh = 32768 + (size_t)TK * 128 * 4 + (size_t)TK * 128 * 2 + 2048;
        fused_scale_kernel<1><<<NSTK, 256, sh, stream>>>(x, xg16, wt[0], cb[0], lg[0], lb[0],
                                                         tl1[0], tl2[0], fcw, fcb, hc, 0);
    }
    {
        constexpr int TK = 64;
        size_t sh = 32768 + (size_t)TK * 128 * 4 + (size_t)TK * 128 * 2 + 2048;
        fused_scale_kernel<2><<<NSTK, 256, sh, stream>>>(x, xg16, wt[1], cb[1], lg[1], lb[1],
                                                         tl1[1], tl2[1], fcw, fcb, hc, 128);
    }
    {
        constexpr int TK = 32;
        size_t sh = 32768 + (size_t)TK * 128 * 4 + (size_t)TK * 128 * 2 + 2048;
        fused_scale_kernel<4><<<NSTK, 256, sh, stream>>>(x, xg16, wt[2], cb[2], lg[2], lb[2],
                                                         tl1[2], tl2[2], fcw, fcb, hc, 192);
    }

    // ---- global LN stats (deterministic two-stage) + stock mixing + head ----
    reduce1<<<RED_BLOCKS, 256, 0, stream>>>(hc, partials);
    reduce2<<<1, 256, 0, stream>>>(partials, stats);
    mix1_kernel<<<MIX, 256, 0, stream>>>(hc, stats, sm_lg, sm_lb, M1, gbuf);
    mix2_kernel<<<NSTK, 256, 0, stream>>>(hc, gbuf, M2, sm_fcw, sm_fcb, out);
}
